// MultiHeadAttention_88630945120379
// MI455X (gfx1250) — matrix operand order repaired and checked by reading, not compiled, not run on hardware
//
#include <hip/hip_runtime.h>
#include <hip/hip_bf16.h>
#include <stdint.h>

// MHA forward: B=4, S=2048, D=1024, H=16, hd=64. All matmuls via
// v_wmma_f32_16x16x32_bf16 (wave32, 16x16 tiles, f32 accum).
// Tile staging uses CDNA5 async LDS loads (ASYNCcnt) and the Tensor Data
// Mover (tensor_load_to_lds, TENSORcnt) for the pure-copy Q tile.

typedef __attribute__((ext_vector_type(16))) __bf16 v16bf;
typedef __attribute__((ext_vector_type(8)))  float  v8f;
typedef unsigned int u32x4 __attribute__((ext_vector_type(4)));
typedef unsigned int u32x8 __attribute__((ext_vector_type(8)));

#define S_LEN 2048
#define D_DIM 1024
#define N_H   16
#define HD    64
#define NTOK  8192   // B*S

union FragAB { v16bf v; uint4 q[2]; };

__device__ __forceinline__ unsigned short f2bf(float f) {
  union { float f; unsigned int u; } v; v.f = f;
  unsigned int r = v.u + 0x7FFFu + ((v.u >> 16) & 1u);   // round-to-nearest-even
  return (unsigned short)(r >> 16);
}

__device__ __forceinline__ v8f wmma_bf16(const FragAB& a, const FragAB& b, v8f c) {
  return __builtin_amdgcn_wmma_f32_16x16x32_bf16(false, a.v, false, b.v,
                                                 (short)0, c, false, false);
}

// CDNA5 async copy: 16 bytes per lane, global -> LDS, tracked by ASYNCcnt.
__device__ __forceinline__ void async_copy_b128(unsigned ldsOff, const void* gsrc) {
  asm volatile("global_load_async_to_lds_b128 %0, %1, off"
               :: "v"(ldsOff), "v"(gsrc) : "memory");
}
__device__ __forceinline__ void wait_async0() {
  asm volatile("s_wait_asynccnt 0x0" ::: "memory");
}
__device__ __forceinline__ void wait_tensor0() {
  asm volatile("s_wait_tensorcnt 0x0" ::: "memory");
}

// Tensor Data Mover: 2-D bf16 tile copy global -> LDS (D# groups 0/1, §8.3/8.4).
// tile_d0 elems per row (dim0, contiguous), tile_d1 rows; stride_d0 in elems.
__device__ __forceinline__ void tdm_load_to_lds_2d(
    unsigned ldsAddr, const void* gsrc,
    unsigned tile_d0, unsigned tile_d1,
    unsigned tensor_d0, unsigned tensor_d1, unsigned stride_d0)
{
  unsigned long long ga = (unsigned long long)gsrc;
  u32x4 g0;
  g0[0] = 1u;                                           // count=1 (valid user D#)
  g0[1] = ldsAddr;                                      // lds_addr
  g0[2] = (unsigned)ga;                                 // global_addr[31:0]
  g0[3] = (unsigned)((ga >> 32) & 0x01FFFFFFu)          // global_addr[56:32]
          | (2u << 30);                                 // type=2 ("image")
  u32x8 g1;
  g1[0] = 1u << 16;                                     // data_size=1 (2B/elem)
  g1[1] = (tensor_d0 & 0xFFFFu) << 16;                  // tensor_dim0 lo16
  g1[2] = (tensor_d0 >> 16) | ((tensor_d1 & 0xFFFFu) << 16);
  g1[3] = (tensor_d1 >> 16) | ((tile_d0 & 0xFFFFu) << 16);   // tile_dim0
  g1[4] = tile_d1 & 0xFFFFu;                            // tile_dim1 (tile_dim2=0)
  g1[5] = stride_d0;                                    // tensor_dim0_stride lo32
  g1[6] = 0;
  g1[7] = 0;
  asm volatile("tensor_load_to_lds %0, %1" :: "s"(g0), "s"(g1) : "memory");
}

// ---------------------------------------------------------------------------
// Kernel 1: QKV projection. Block tile 128 rows x 128 cols (2 heads), 8 waves,
// K-step 64 -> 16 WMMAs per wave per step.
// out[b,h,s,e] = (X[b,s,:] @ W[h,:,e] + bias[h,e]) * scale
// ---------------------------------------------------------------------------
__global__ __launch_bounds__(256) void proj_kernel(
    const float* __restrict__ X,          // [NTOK, D]
    const float* __restrict__ W,          // [H, D, HD]
    const float* __restrict__ bias,       // [H*HD] flat
    unsigned short* __restrict__ out,     // [B, H, S, HD] bf16
    float scale)
{
  __shared__ __align__(16) unsigned short As[128 * 64];
  __shared__ __align__(16) unsigned short Wt[128 * 64];  // transposed [c][k]

  const int tid  = threadIdx.x;
  const int w    = tid >> 5;
  const int lane = tid & 31;
  const int lr   = lane & 15;
  const int hb   = (lane < 16) ? 0 : 8;    // A-fragment K base
  const int kb   = (lane < 16) ? 0 : 16;   // unused; B fragment also uses hb
  const int hb8  = (lane < 16) ? 0 : 8;    // C/D row offset

  const int row0 = blockIdx.x * 128;
  const int nc0  = blockIdx.y * 128;       // column block over H*HD space

  v8f zero = {0,0,0,0,0,0,0,0};
  v8f acc[8];
#pragma unroll
  for (int n = 0; n < 8; ++n) acc[n] = zero;

  for (int k0 = 0; k0 < D_DIM; k0 += 64) {
    // Stage A tile 128x64 (f32 -> bf16)
#pragma unroll
    for (int j = 0; j < 8; ++j) {
      int idx = tid + j * 256;              // 2048 float4 units
      int r   = idx >> 4, c4 = idx & 15;
      float4 f = ((const float4*)(X + (size_t)(row0 + r) * D_DIM + k0))[c4];
      uint2 p;
      p.x = (unsigned)f2bf(f.x) | ((unsigned)f2bf(f.y) << 16);
      p.y = (unsigned)f2bf(f.z) | ((unsigned)f2bf(f.w) << 16);
      *(uint2*)&As[r * 64 + c4 * 4] = p;
    }
    // Stage W tile transposed: Wt[c][kk], columns c cover 2 heads
#pragma unroll
    for (int j = 0; j < 32; ++j) {
      int idx = tid + j * 256;              // 8192 elements
      int kk = idx >> 7, c = idx & 127;
      int gc = nc0 + c;
      int h = gc >> 6, e = gc & 63;
      float wv = W[(size_t)h * D_DIM * HD + (size_t)(k0 + kk) * HD + e];
      Wt[c * 64 + kk] = f2bf(wv);
    }
    __syncthreads();

#pragma unroll
    for (int ek = 0; ek < 64; ek += 32) {
      FragAB a;
      a.q[0] = *(const uint4*)&As[(w * 16 + lr) * 64 + ek + hb];
      a.q[1] = *(const uint4*)&As[(w * 16 + lr) * 64 + ek + hb + 16];
#pragma unroll
      for (int n = 0; n < 8; ++n) {
        FragAB bf;
        bf.q[0] = *(const uint4*)&Wt[(n * 16 + lr) * 64 + ek + hb];
        bf.q[1] = *(const uint4*)&Wt[(n * 16 + lr) * 64 + ek + hb + 16];
        acc[n] = wmma_bf16(a, bf, acc[n]);
      }
    }
    __syncthreads();
  }

#pragma unroll
  for (int n = 0; n < 8; ++n) {
    const int c  = nc0 + n * 16 + lr;       // flat column = h*64+e
    const int h  = c >> 6, e = c & 63;
    const float bv = bias[c];
#pragma unroll
    for (int r = 0; r < 8; ++r) {
      int grow = row0 + w * 16 + r + hb8;
      int bidx = grow >> 11;               // / S_LEN
      int s    = grow & (S_LEN - 1);
      float val = (acc[n][r] + bv) * scale;
      out[(((size_t)(bidx * N_H + h)) * S_LEN + s) * HD + e] = f2bf(val);
    }
  }
}

// ---------------------------------------------------------------------------
// Kernel 2: causal flash attention per (b, h, 128-query tile).
// Q tile via TDM, K tiles via async LDS loads, V transposed manually.
// ---------------------------------------------------------------------------
__global__ __launch_bounds__(256) void attn_kernel(
    const unsigned short* __restrict__ qb,   // [B,H,S,HD] bf16 (pre-scaled)
    const unsigned short* __restrict__ kb,
    const unsigned short* __restrict__ vb,
    unsigned short* __restrict__ ob)         // [B*S, D] bf16 (heads concat)
{
  __shared__ __align__(16) unsigned short Qs[128 * 64];       // [m][e]
  __shared__ __align__(16) unsigned short Ks[128 * 64];       // [t][e]
  __shared__ __align__(16) unsigned short Vs[64 * 128];       // [e][t] transposed
  __shared__ __align__(16) unsigned short Ps[8 * 16 * 128];   // per-wave P slab

  const int tid  = threadIdx.x;
  const int w    = tid >> 5;
  const int lane = tid & 31;
  const int lr   = lane & 15;
  const int hb   = (lane < 16) ? 0 : 8;
  const int kbse = (lane < 16) ? 0 : 16;
  const int hb8  = hb;

  const int qt = blockIdx.x;
  const int h  = blockIdx.y;
  const int b  = blockIdx.z;
  const size_t hbase = ((size_t)(b * N_H + h)) * S_LEN * HD;

  // TDM: DMA the 128x64 bf16 Q tile straight into LDS (wave 0 only; EXEC is
  // ignored by tensor ops so one wave issues exactly one DMA).
  if (w == 0) {
    tdm_load_to_lds_2d((unsigned)(size_t)&Qs[0],
                       qb + hbase + (size_t)qt * 128 * HD,
                       /*tile_d0=*/HD, /*tile_d1=*/128,
                       /*tensor_d0=*/HD, /*tensor_d1=*/S_LEN,
                       /*stride_d0=*/HD);
  }

  v8f zero = {0,0,0,0,0,0,0,0};
  v8f oacc[4];
  float m[8], l[8];
#pragma unroll
  for (int n = 0; n < 4; ++n) oacc[n] = zero;
#pragma unroll
  for (int r = 0; r < 8; ++r) { m[r] = -3.0e38f; l[r] = 0.0f; }

  for (int kt = 0; kt <= qt; ++kt) {
    const int t0 = kt * 128;
    // Async-stage K tile [t][e]
    {
      const unsigned short* src = kb + hbase + (size_t)t0 * HD;
#pragma unroll
      for (int j = 0; j < 4; ++j) {
        int idx = tid + j * 256;
        async_copy_b128((unsigned)(size_t)&Ks[idx * 8], src + idx * 8);
      }
    }
    // Stage V tile transposed [e][t] (needs transpose, so via VGPRs)
    const uint4* vsrc = (const uint4*)(vb + hbase + (size_t)t0 * HD);
#pragma unroll
    for (int j = 0; j < 4; ++j) {
      int idx = tid + j * 256;              // 1024 uint4 units, 8 per row
      int t = idx >> 3, e8 = (idx & 7) * 8;
      union { uint4 q; unsigned short s[8]; } u;
      u.q = vsrc[idx];
#pragma unroll
      for (int i = 0; i < 8; ++i) Vs[(e8 + i) * 128 + t] = u.s[i];
    }
    if (kt < qt)  // prefetch next K tile into GL2
      __builtin_prefetch(kb + hbase + (size_t)(t0 + 128) * HD, 0, 1);
    if (kt == 0 && w == 0) wait_tensor0();   // Q tile DMA complete
    wait_async0();
    __syncthreads();

    // --- scores: S = Q (16xHD) x K^T (HD x 128), per wave ---
    v8f sacc[8];
#pragma unroll
    for (int n = 0; n < 8; ++n) sacc[n] = zero;
#pragma unroll
    for (int ek = 0; ek < HD; ek += 32) {
      FragAB a;
      a.q[0] = *(const uint4*)&Qs[(w * 16 + lr) * HD + ek + hb];
      a.q[1] = *(const uint4*)&Qs[(w * 16 + lr) * HD + ek + hb + 16];
#pragma unroll
      for (int n = 0; n < 8; ++n) {
        FragAB bf;
        bf.q[0] = *(const uint4*)&Ks[(n * 16 + lr) * HD + ek + hb];
        bf.q[1] = *(const uint4*)&Ks[(n * 16 + lr) * HD + ek + hb + 16];
        sacc[n] = wmma_bf16(a, bf, sacc[n]);
      }
    }

    // Causal mask only on the diagonal tile
    if (kt == qt) {
#pragma unroll
      for (int n = 0; n < 8; ++n) {
        int jg = t0 + n * 16 + lr;
#pragma unroll
        for (int r = 0; r < 8; ++r) {
          int ig = qt * 128 + w * 16 + r + hb8;
          if (jg > ig) sacc[n][r] = -3.0e38f;
        }
      }
    }

    // --- online softmax (rows live in VGPR index r, cols across 16 lanes) ---
    float tm[8];
#pragma unroll
    for (int r = 0; r < 8; ++r) {
      float v = sacc[0][r];
#pragma unroll
      for (int n = 1; n < 8; ++n) v = fmaxf(v, sacc[n][r]);
      tm[r] = v;
    }
#pragma unroll
    for (int r = 0; r < 8; ++r)
#pragma unroll
      for (int off = 1; off < 16; off <<= 1)
        tm[r] = fmaxf(tm[r], __shfl_xor(tm[r], off, 32));

#pragma unroll
    for (int r = 0; r < 8; ++r) {
      float nm  = fmaxf(m[r], tm[r]);
      float scl = __expf(m[r] - nm);
      m[r] = nm;
      l[r] *= scl;
#pragma unroll
      for (int n = 0; n < 4; ++n) oacc[n][r] *= scl;
    }

    float rs[8];
#pragma unroll
    for (int r = 0; r < 8; ++r) rs[r] = 0.0f;
#pragma unroll
    for (int n = 0; n < 8; ++n)
#pragma unroll
      for (int r = 0; r < 8; ++r) {
        float p = __expf(sacc[n][r] - m[r]);
        rs[r] += p;
        sacc[n][r] = p;
      }
#pragma unroll
    for (int r = 0; r < 8; ++r) {
#pragma unroll
      for (int off = 1; off < 16; off <<= 1)
        rs[r] += __shfl_xor(rs[r], off, 32);
      l[r] += rs[r];
    }

    // Write P (C-layout) to per-wave LDS slab as bf16 (becomes A-layout source)
    unsigned short* Pw = &Ps[w * 16 * 128];
#pragma unroll
    for (int n = 0; n < 8; ++n)
#pragma unroll
      for (int r = 0; r < 8; ++r)
        Pw[(r + hb8) * 128 + n * 16 + lr] = f2bf(sacc[n][r]);

    // --- O += P (16x128) x V (128xHD) ---
#pragma unroll
    for (int ek2 = 0; ek2 < 128; ek2 += 32) {
      FragAB a;
      a.q[0] = *(const uint4*)&Pw[lr * 128 + ek2 + hb];
      a.q[1] = *(const uint4*)&Pw[lr * 128 + ek2 + hb + 16];
#pragma unroll
      for (int n = 0; n < 4; ++n) {
        FragAB bf;
        bf.q[0] = *(const uint4*)&Vs[(n * 16 + lr) * 128 + ek2 + hb];
        bf.q[1] = *(const uint4*)&Vs[(n * 16 + lr) * 128 + ek2 + hb + 16];
        oacc[n] = wmma_bf16(a, bf, oacc[n]);
      }
    }
    __syncthreads();
  }

  // Normalize and store concat-head output [b*S+s][h*64+e] bf16
#pragma unroll
  for (int r = 0; r < 8; ++r) {
    int s = qt * 128 + w * 16 + r + hb8;
    float inv = 1.0f / l[r];
#pragma unroll
    for (int n = 0; n < 4; ++n) {
      float val = oacc[n][r] * inv;
      ob[((size_t)(b * S_LEN + s)) * D_DIM + h * HD + n * 16 + lr] = f2bf(val);
    }
  }
}

// ---------------------------------------------------------------------------
// Kernel 3: output projection  out = O @ Wo + bo  (f32 result).
// Block tile 128x128, K-step 64; A tile staged with async LDS loads.
// ---------------------------------------------------------------------------
__global__ __launch_bounds__(256) void outproj_kernel(
    const unsigned short* __restrict__ ob,  // [NTOK, D] bf16
    const float* __restrict__ Wo,           // [D, D]
    const float* __restrict__ bo,           // [D]
    float* __restrict__ out)                // [NTOK, D]
{
  __shared__ __align__(16) unsigned short As[128 * 64];
  __shared__ __align__(16) unsigned short Wt[128 * 64];

  const int tid  = threadIdx.x;
  const int w    = tid >> 5;
  const int lane = tid & 31;
  const int lr   = lane & 15;
  const int hb   = (lane < 16) ? 0 : 8;
  const int kb   = (lane < 16) ? 0 : 16;
  const int hb8  = (lane < 16) ? 0 : 8;

  const int row0 = blockIdx.x * 128;
  const int nc0  = blockIdx.y * 128;

  v8f zero = {0,0,0,0,0,0,0,0};
  v8f acc[8];
#pragma unroll
  for (int n = 0; n < 8; ++n) acc[n] = zero;

  for (int k0 = 0; k0 < D_DIM; k0 += 64) {
    // Async-stage A tile 128x64 bf16 (no conversion needed)
#pragma unroll
    for (int j = 0; j < 4; ++j) {
      int idx = tid + j * 256;              // 1024 uint4 units
      int r = idx >> 3, c = idx & 7;
      async_copy_b128((unsigned)(size_t)&As[r * 64 + c * 8],
                      ob + (size_t)(row0 + r) * D_DIM + k0 + c * 8);
    }
    // Stage Wo tile transposed (f32 -> bf16): Wt[c][kk]
#pragma unroll
    for (int j = 0; j < 32; ++j) {
      int idx = tid + j * 256;              // 8192 elements
      int kk = idx >> 7, c = idx & 127;
      Wt[c * 64 + kk] = f2bf(Wo[(size_t)(k0 + kk) * D_DIM + nc0 + c]);
    }
    wait_async0();
    __syncthreads();

#pragma unroll
    for (int ek = 0; ek < 64; ek += 32) {
      FragAB a;
      a.q[0] = *(const uint4*)&As[(w * 16 + lr) * 64 + ek + hb];
      a.q[1] = *(const uint4*)&As[(w * 16 + lr) * 64 + ek + hb + 16];
#pragma unroll
      for (int n = 0; n < 8; ++n) {
        FragAB bf;
        bf.q[0] = *(const uint4*)&Wt[(n * 16 + lr) * 64 + ek + hb];
        bf.q[1] = *(const uint4*)&Wt[(n * 16 + lr) * 64 + ek + hb + 16];
        acc[n] = wmma_bf16(a, bf, acc[n]);
      }
    }
    __syncthreads();
  }

#pragma unroll
  for (int n = 0; n < 8; ++n) {
    int col = nc0 + n * 16 + lr;
    float bv = bo[col];
#pragma unroll
    for (int r = 0; r < 8; ++r) {
      int grow = row0 + w * 16 + r + hb8;
      out[(size_t)grow * D_DIM + col] = acc[n][r] + bv;
    }
  }
}

// ---------------------------------------------------------------------------
extern "C" void kernel_launch(void* const* d_in, const int* in_sizes, int n_in,
                              void* d_out, int out_size, void* d_ws, size_t ws_size,
                              hipStream_t stream) {
  const float* keys    = (const float*)d_in[0];
  const float* queries = (const float*)d_in[1];
  const float* values  = (const float*)d_in[2];
  const float* Wq      = (const float*)d_in[3];
  const float* bq      = (const float*)d_in[4];
  const float* Wk      = (const float*)d_in[5];
  const float* bk      = (const float*)d_in[6];
  const float* Wv      = (const float*)d_in[7];
  const float* bv      = (const float*)d_in[8];
  const float* Wo      = (const float*)d_in[9];
  const float* bo      = (const float*)d_in[10];

  // Workspace: q, k, v [B,H,S,HD] bf16 + o [NTOK,D] bf16 = 64 MB total
  const size_t PLANE = (size_t)NTOK * D_DIM;   // elements
  unsigned short* qbuf = (unsigned short*)d_ws;
  unsigned short* kbuf = qbuf + PLANE;
  unsigned short* vbuf = kbuf + PLANE;
  unsigned short* obuf = vbuf + PLANE;

  dim3 blk(256);
  // Q scaled by 1/sqrt(hd)=0.125 so QK^T needs no extra scale.
  proj_kernel<<<dim3(NTOK / 128, 8), blk, 0, stream>>>(queries, Wq, bq, qbuf, 0.125f);
  proj_kernel<<<dim3(NTOK / 128, 8), blk, 0, stream>>>(keys,    Wk, bk, kbuf, 1.0f);
  proj_kernel<<<dim3(NTOK / 128, 8), blk, 0, stream>>>(values,  Wv, bv, vbuf, 1.0f);

  attn_kernel<<<dim3(S_LEN / 128, N_H, 4), blk, 0, stream>>>(qbuf, kbuf, vbuf, obuf);

  outproj_kernel<<<dim3(NTOK / 128, 8), blk, 0, stream>>>(obuf, Wo, bo,
                                                          (float*)d_out);
}
